// AttentionMechanism_23639499997673
// MI455X (gfx1250) — matrix operand order repaired and checked by reading, not compiled, not run on hardware
//
#include <hip/hip_runtime.h>

// Problem constants
#define B_ 4
#define L_ 1024
#define D_ 768
#define H_ 12
#define HD_ 64
#define MAXLEN_ 1024

typedef __attribute__((ext_vector_type(16))) _Float16 v16h;
typedef __attribute__((ext_vector_type(8)))  _Float16 v8h;
typedef __attribute__((ext_vector_type(8)))  float    v8f;
typedef __attribute__((ext_vector_type(4)))  unsigned int v4u;
typedef __attribute__((ext_vector_type(4)))  int v4i;
typedef __attribute__((ext_vector_type(8)))  int v8i;

#if defined(__has_builtin)
#if __has_builtin(__builtin_amdgcn_tensor_load_to_lds)
#define HAVE_TDM 1
#endif
#endif

// ---------------------------------------------------------------------------
// WMMA fragment loaders (wave32, 16x16x32 f16 -> f32)
// A layout: lanes0-15 row=lane  K0..15 contiguous; lanes16-31 row=lane-16 K16..31
// B layout: lanes0-15 col=lane  K0..15 contiguous; lanes16-31 K16..31
// All B operands are stored transposed ([N][K] row-major).
// ---------------------------------------------------------------------------
__device__ __forceinline__ v16h load_a_frag(const _Float16* base, int row,
                                            int stride_h, int k0, int lane) {
  const _Float16* p = base + (size_t)row * stride_h + k0 + ((lane >> 4) << 4);
  v16h r;
  ((v8h*)&r)[0] = *(const v8h*)(p);
  ((v8h*)&r)[1] = *(const v8h*)(p + 8);
  return r;
}

__device__ __forceinline__ v16h load_b_frag(const _Float16* base, int row,
                                            int stride_h, int k0, int lane) {
  const _Float16* p = base + (size_t)row * stride_h + k0 + ((lane >> 4) << 4);
  return *(const v16h*)(p);
}

__device__ __forceinline__ v8f wmma_f16(v16h a, v16h b, v8f c) {
  return __builtin_amdgcn_wmma_f32_16x16x32_f16(false, a, false, b, (short)0, c,
                                                false, false);
}

#ifdef HAVE_TDM
// Issue a TDM 2-D tile load (f16 elements) into LDS. All args wave-uniform.
// D# packing per CDNA5 ISA 8.3/8.4; dims/strides in element units.
__device__ __forceinline__ void tdm_load_2d(unsigned lds_off, const _Float16* gptr,
                                            unsigned tensor_d0, unsigned tensor_d1,
                                            unsigned tile_d0, unsigned tile_d1,
                                            unsigned stride0) {
  unsigned long long ga = (unsigned long long)(uintptr_t)gptr;
  v4u g0;
  g0[0] = 1u;                                   // count=1 valid descriptor
  g0[1] = lds_off;                              // lds_addr (bytes)
  g0[2] = (unsigned)ga;                         // global_addr[31:0]
  g0[3] = (unsigned)((ga >> 32) & 0x1FFFFFFu) | (2u << 30);  // addr[56:32], type=2
  v8i g1;
  g1[0] = (int)(1u << 16);                      // data_size=2B; wg_mask=0
  g1[1] = (int)((tensor_d0 & 0xFFFFu) << 16);   // tensor_dim0[15:0]
  g1[2] = (int)((tensor_d0 >> 16) | ((tensor_d1 & 0xFFFFu) << 16));
  g1[3] = (int)((tensor_d1 >> 16) | (tile_d0 << 16));
  g1[4] = (int)(tile_d1 & 0xFFFFu);             // tile_dim1; tile_dim2=0
  g1[5] = (int)stride0;                         // tensor_dim0_stride[31:0]
  g1[6] = 0;                                    // stride0 hi / stride1 lo
  g1[7] = 0;
  v4i z4 = {0, 0, 0, 0};
#if __clang_major__ >= 23
  v8i z8 = {0, 0, 0, 0, 0, 0, 0, 0};
  __builtin_amdgcn_tensor_load_to_lds(g0, g1, z4, z4, z8, 0);
#else
  __builtin_amdgcn_tensor_load_to_lds(g0, g1, z4, z4, 0);
#endif
}
#endif

// ---------------------------------------------------------------------------
// Conversion / transpose kernels
// ---------------------------------------------------------------------------
__global__ void k_f32_to_f16(const float* __restrict__ in,
                             _Float16* __restrict__ out, int n) {
  int i = blockIdx.x * 256 + threadIdx.x;
  if (i < n) out[i] = (_Float16)in[i];
}

__global__ void k_transpose_f16(const float* __restrict__ W,
                                _Float16* __restrict__ Wt) {
  int k = blockIdx.x * 16 + threadIdx.x;
  int n = blockIdx.y * 16 + threadIdx.y;
  Wt[(size_t)n * D_ + k] = (_Float16)W[(size_t)k * D_ + n];
}

// ---------------------------------------------------------------------------
// QKV projection: one wave computes a 32(M) x 64(N) tile of X @ W + b with
// register double-buffering (next k-step fragments loaded before current WMMAs).
// vmode 0 -> write [B,H,L,HD] (q,k); vmode 1 -> write [B,H,HD,L] (v^T).
// ---------------------------------------------------------------------------
__global__ void __launch_bounds__(32) k_proj_qkv(const _Float16* __restrict__ Xh,
                                                 const _Float16* __restrict__ Wt,
                                                 const float* __restrict__ bias,
                                                 _Float16* __restrict__ out,
                                                 int vmode) {
  const int lane = threadIdx.x & 31;
  const int m0 = blockIdx.x * 32;
  const int n0 = blockIdx.y * 64;
  const int half = lane >> 4, nl = lane & 15;

  v8f acc[2][4];
  for (int mi = 0; mi < 2; ++mi)
    for (int t = 0; t < 4; ++t)
      for (int v = 0; v < 8; ++v) acc[mi][t][v] = 0.f;

  v16h a0 = load_a_frag(Xh, m0 + nl, D_, 0, lane);
  v16h a1 = load_a_frag(Xh, m0 + 16 + nl, D_, 0, lane);
  v16h b0 = load_b_frag(Wt, n0 + 0 + nl, D_, 0, lane);
  v16h b1 = load_b_frag(Wt, n0 + 16 + nl, D_, 0, lane);
  v16h b2 = load_b_frag(Wt, n0 + 32 + nl, D_, 0, lane);
  v16h b3 = load_b_frag(Wt, n0 + 48 + nl, D_, 0, lane);

  for (int k = 0; k < D_; k += 32) {
    const int kn = (k + 32 < D_) ? k + 32 : 0;   // branch-free prefetch index
    v16h na0 = load_a_frag(Xh, m0 + nl, D_, kn, lane);
    v16h na1 = load_a_frag(Xh, m0 + 16 + nl, D_, kn, lane);
    v16h nb0 = load_b_frag(Wt, n0 + 0 + nl, D_, kn, lane);
    v16h nb1 = load_b_frag(Wt, n0 + 16 + nl, D_, kn, lane);
    v16h nb2 = load_b_frag(Wt, n0 + 32 + nl, D_, kn, lane);
    v16h nb3 = load_b_frag(Wt, n0 + 48 + nl, D_, kn, lane);

    acc[0][0] = wmma_f16(a0, b0, acc[0][0]);
    acc[0][1] = wmma_f16(a0, b1, acc[0][1]);
    acc[0][2] = wmma_f16(a0, b2, acc[0][2]);
    acc[0][3] = wmma_f16(a0, b3, acc[0][3]);
    acc[1][0] = wmma_f16(a1, b0, acc[1][0]);
    acc[1][1] = wmma_f16(a1, b1, acc[1][1]);
    acc[1][2] = wmma_f16(a1, b2, acc[1][2]);
    acc[1][3] = wmma_f16(a1, b3, acc[1][3]);

    a0 = na0; a1 = na1; b0 = nb0; b1 = nb1; b2 = nb2; b3 = nb3;
  }

#pragma unroll
  for (int mi = 0; mi < 2; ++mi)
#pragma unroll
    for (int t = 0; t < 4; ++t) {
      int n = n0 + t * 16 + nl;
      float bv = bias[n];
      int h = n >> 6, d = n & 63;
#pragma unroll
      for (int v = 0; v < 8; ++v) {
        int m = m0 + mi * 16 + v + half * 8;     // C layout: row = v + 8*half
        int bb = m >> 10, l = m & 1023;
        float val = acc[mi][t][v] + bv;
        size_t idx = vmode ? ((size_t)((bb * H_ + h) * HD_ + d) * L_ + l)
                           : ((size_t)((bb * H_ + h) * L_ + l) * HD_ + d);
        out[idx] = (_Float16)val;
      }
    }
}

// ---------------------------------------------------------------------------
// Fused relative-position flash attention.
// One wave owns (b, h, 16 query rows); loops r in steps of 32.
// K(32x64) and V^T(64x32) tiles are staged into per-wave LDS double buffers by
// the Tensor Data Mover one step ahead (s_wait_tensorcnt pipelining).
// Per 16-col subtile: scores = Q K^T (2 wmma), RQ = Q Eb^T (4), RK = K Eb^T (4),
// diagonal gather of RQ/RK through LDS, online softmax, ctx += P V (4 wmma).
// ---------------------------------------------------------------------------
__global__ void __launch_bounds__(128) k_attention(
    const _Float16* __restrict__ qh, const _Float16* __restrict__ kh,
    const _Float16* __restrict__ vT, const _Float16* __restrict__ Eh,
    _Float16* __restrict__ ctxh) {
#ifdef HAVE_TDM
  __shared__ alignas(16) _Float16 s_kbuf[4][2][32 * 64];
  __shared__ alignas(16) _Float16 s_vbuf[4][2][64 * 32];
#endif
  __shared__ alignas(16) float    s_rq[4][16 * 32];
  __shared__ alignas(16) float    s_rk[4][16 * 32];
  __shared__ alignas(16) _Float16 s_p[4][16 * 32];

  const int lane = threadIdx.x & 31;
  const int w = threadIdx.x >> 5;
  const int gw = blockIdx.x * 4 + w;          // 3072 waves total
  const int l0 = (gw & 63) * 16;
  const int h  = (gw >> 6) % H_;
  const int b  = gw / (64 * H_);
  const int half = lane >> 4, nl = lane & 15;

  const _Float16* qB = qh + (size_t)(b * H_ + h) * L_ * HD_;
  const _Float16* kB = kh + (size_t)(b * H_ + h) * L_ * HD_;
  const _Float16* vB = vT + (size_t)(b * H_ + h) * HD_ * L_;
  float* rq = s_rq[w];
  float* rk = s_rk[w];
  _Float16* pp = s_p[w];

  v16h qa0 = load_a_frag(qB, l0 + nl, HD_, 0, lane);
  v16h qa1 = load_a_frag(qB, l0 + nl, HD_, 32, lane);

  float mrun[8], lrun[8];
  v8f ctx[4];
  for (int v = 0; v < 8; ++v) { mrun[v] = -3.0e38f; lrun[v] = 0.f; }
  for (int t = 0; t < 4; ++t)
    for (int v = 0; v < 8; ++v) ctx[t][v] = 0.f;

#ifdef HAVE_TDM
  unsigned koff[2], voff[2];
  koff[0] = (unsigned)(uintptr_t)(void*)&s_kbuf[w][0][0];
  koff[1] = (unsigned)(uintptr_t)(void*)&s_kbuf[w][1][0];
  voff[0] = (unsigned)(uintptr_t)(void*)&s_vbuf[w][0][0];
  voff[1] = (unsigned)(uintptr_t)(void*)&s_vbuf[w][1][0];
  // preload first K/V tiles into buffer 0
  tdm_load_2d(koff[0], kB, /*td0*/64, /*td1*/32, /*tile*/64, 32, /*stride0*/64);
  tdm_load_2d(voff[0], vB, /*td0*/32, /*td1*/64, /*tile*/32, 64, /*stride0*/1024);
#endif

  for (int rs = 0; rs < L_; rs += 32) {
#ifdef HAVE_TDM
    const int bufn = (rs >> 5) & 1;
    const _Float16* kcur = &s_kbuf[w][bufn][0];
    const _Float16* vcur = &s_vbuf[w][bufn][0];
    if (rs + 32 < L_) {
      asm volatile("s_wait_dscnt 0" ::: "memory");  // old reads done before overwrite
      tdm_load_2d(koff[bufn ^ 1], kB + (size_t)(rs + 32) * HD_, 64, 32, 64, 32, 64);
      tdm_load_2d(voff[bufn ^ 1], vB + (rs + 32), 32, 64, 32, 64, 1024);
      __builtin_amdgcn_s_wait_tensorcnt(2);         // current buffers ready
    } else {
      __builtin_amdgcn_s_wait_tensorcnt(0);
    }
#endif
#pragma unroll
    for (int jt = 0; jt < 2; ++jt) {
      const int r0 = rs + jt * 16;
#ifdef HAVE_TDM
      v16h kb0 = load_b_frag(kcur, jt * 16 + nl, HD_, 0, lane);
      v16h kb1 = load_b_frag(kcur, jt * 16 + nl, HD_, 32, lane);
      v16h ka0 = load_a_frag(kcur, jt * 16 + nl, HD_, 0, lane);
      v16h ka1 = load_a_frag(kcur, jt * 16 + nl, HD_, 32, lane);
#else
      v16h kb0 = load_b_frag(kB, r0 + nl, HD_, 0, lane);
      v16h kb1 = load_b_frag(kB, r0 + nl, HD_, 32, lane);
      v16h ka0 = load_a_frag(kB, r0 + nl, HD_, 0, lane);
      v16h ka1 = load_a_frag(kB, r0 + nl, HD_, 32, lane);
#endif
      v8f s;
      for (int v = 0; v < 8; ++v) s[v] = 0.f;
      s = wmma_f16(qa0, kb0, s);
      s = wmma_f16(qa1, kb1, s);

      // Toeplitz rel terms: 31 emb rows starting at dmin; (i,jr) -> j = i-jr+15
      const int dmin = l0 - r0 + 1008;
      v8f accq[2], acck[2];
#pragma unroll
      for (int t = 0; t < 2; ++t) {
        int er = dmin + t * 16 + nl;
        er = er < 0 ? 0 : (er > 2046 ? 2046 : er);
        v16h e0 = load_b_frag(Eh, er, HD_, 0, lane);
        v16h e1 = load_b_frag(Eh, er, HD_, 32, lane);
        v8f aq, ak;
        for (int v = 0; v < 8; ++v) { aq[v] = 0.f; ak[v] = 0.f; }
        aq = wmma_f16(qa0, e0, aq);
        aq = wmma_f16(qa1, e1, aq);
        ak = wmma_f16(ka0, e0, ak);
        ak = wmma_f16(ka1, e1, ak);
        accq[t] = aq;
        acck[t] = ak;
      }
#pragma unroll
      for (int t = 0; t < 2; ++t)
#pragma unroll
        for (int v = 0; v < 8; ++v) {
          int row = v + half * 8;
          rq[row * 32 + t * 16 + nl] = accq[t][v];
          rk[row * 32 + t * 16 + nl] = acck[t][v];
        }
      asm volatile("s_wait_dscnt 0" ::: "memory");

#pragma unroll
      for (int v = 0; v < 8; ++v) {
        int i = v + half * 8;
        int j = i - nl + 15;                    // 0..30
        float sv = (s[v] + rq[i * 32 + j] + rk[nl * 32 + j]) * 0.125f;
        float rmax = sv;
        rmax = fmaxf(rmax, __shfl_xor(rmax, 1, 32));
        rmax = fmaxf(rmax, __shfl_xor(rmax, 2, 32));
        rmax = fmaxf(rmax, __shfl_xor(rmax, 4, 32));
        rmax = fmaxf(rmax, __shfl_xor(rmax, 8, 32));
        float newm = fmaxf(mrun[v], rmax);
        float scale = __expf(mrun[v] - newm);
        float p = __expf(sv - newm);
        float rsum = p;
        rsum += __shfl_xor(rsum, 1, 32);
        rsum += __shfl_xor(rsum, 2, 32);
        rsum += __shfl_xor(rsum, 4, 32);
        rsum += __shfl_xor(rsum, 8, 32);
        lrun[v] = lrun[v] * scale + rsum;
        mrun[v] = newm;
#pragma unroll
        for (int t = 0; t < 4; ++t) ctx[t][v] *= scale;
        pp[i * 32 + jt * 16 + nl] = (_Float16)p;
      }
    }
    asm volatile("s_wait_dscnt 0" ::: "memory");
    v16h pa = load_a_frag(pp, nl, 32, 0, lane);
#pragma unroll
    for (int t = 0; t < 4; ++t) {
#ifdef HAVE_TDM
      v16h vop = load_b_frag(vcur, t * 16 + nl, 32, 0, lane);
#else
      v16h vop = load_b_frag(vB, t * 16 + nl, L_, rs, lane);
#endif
      ctx[t] = wmma_f16(pa, vop, ctx[t]);
    }
  }

#pragma unroll
  for (int t = 0; t < 4; ++t)
#pragma unroll
    for (int v = 0; v < 8; ++v) {
      int i = v + half * 8;
      float val = ctx[t][v] / lrun[v];
      size_t m = (size_t)b * L_ + l0 + i;
      ctxh[m * D_ + h * HD_ + t * 16 + nl] = (_Float16)val;
    }
}

// ---------------------------------------------------------------------------
// Output projection: out = ctx @ Wd + bd (f32 out), 32x64 tile, double-buffered
// ---------------------------------------------------------------------------
__global__ void __launch_bounds__(32) k_out_proj(const _Float16* __restrict__ Ah,
                                                 const _Float16* __restrict__ Wt,
                                                 const float* __restrict__ bias,
                                                 float* __restrict__ out) {
  const int lane = threadIdx.x & 31;
  const int m0 = blockIdx.x * 32;
  const int n0 = blockIdx.y * 64;
  const int half = lane >> 4, nl = lane & 15;

  v8f acc[2][4];
  for (int mi = 0; mi < 2; ++mi)
    for (int t = 0; t < 4; ++t)
      for (int v = 0; v < 8; ++v) acc[mi][t][v] = 0.f;

  v16h a0 = load_a_frag(Ah, m0 + nl, D_, 0, lane);
  v16h a1 = load_a_frag(Ah, m0 + 16 + nl, D_, 0, lane);
  v16h b0 = load_b_frag(Wt, n0 + 0 + nl, D_, 0, lane);
  v16h b1 = load_b_frag(Wt, n0 + 16 + nl, D_, 0, lane);
  v16h b2 = load_b_frag(Wt, n0 + 32 + nl, D_, 0, lane);
  v16h b3 = load_b_frag(Wt, n0 + 48 + nl, D_, 0, lane);

  for (int k = 0; k < D_; k += 32) {
    const int kn = (k + 32 < D_) ? k + 32 : 0;
    v16h na0 = load_a_frag(Ah, m0 + nl, D_, kn, lane);
    v16h na1 = load_a_frag(Ah, m0 + 16 + nl, D_, kn, lane);
    v16h nb0 = load_b_frag(Wt, n0 + 0 + nl, D_, kn, lane);
    v16h nb1 = load_b_frag(Wt, n0 + 16 + nl, D_, kn, lane);
    v16h nb2 = load_b_frag(Wt, n0 + 32 + nl, D_, kn, lane);
    v16h nb3 = load_b_frag(Wt, n0 + 48 + nl, D_, kn, lane);

    acc[0][0] = wmma_f16(a0, b0, acc[0][0]);
    acc[0][1] = wmma_f16(a0, b1, acc[0][1]);
    acc[0][2] = wmma_f16(a0, b2, acc[0][2]);
    acc[0][3] = wmma_f16(a0, b3, acc[0][3]);
    acc[1][0] = wmma_f16(a1, b0, acc[1][0]);
    acc[1][1] = wmma_f16(a1, b1, acc[1][1]);
    acc[1][2] = wmma_f16(a1, b2, acc[1][2]);
    acc[1][3] = wmma_f16(a1, b3, acc[1][3]);

    a0 = na0; a1 = na1; b0 = nb0; b1 = nb1; b2 = nb2; b3 = nb3;
  }

#pragma unroll
  for (int mi = 0; mi < 2; ++mi)
#pragma unroll
    for (int t = 0; t < 4; ++t) {
      int n = n0 + t * 16 + nl;
      float bv = bias[n];
#pragma unroll
      for (int v = 0; v < 8; ++v) {
        int m = m0 + mi * 16 + v + half * 8;
        out[(size_t)m * D_ + n] = acc[mi][t][v] + bv;
      }
    }
}

// ---------------------------------------------------------------------------
extern "C" void kernel_launch(void* const* d_in, const int* in_sizes, int n_in,
                              void* d_out, int out_size, void* d_ws, size_t ws_size,
                              hipStream_t stream) {
  (void)in_sizes; (void)n_in; (void)out_size; (void)ws_size;
  const float* hidden = (const float*)d_in[0];
  const float* Wq = (const float*)d_in[1];
  const float* bq = (const float*)d_in[2];
  const float* Wk = (const float*)d_in[3];
  const float* bk = (const float*)d_in[4];
  const float* Wv = (const float*)d_in[5];
  const float* bv = (const float*)d_in[6];
  const float* Wd = (const float*)d_in[7];
  const float* bd = (const float*)d_in[8];
  const float* emb = (const float*)d_in[9];
  float* out = (float*)d_out;

  char* ws = (char*)d_ws;
  size_t off = 0;
  auto alloc = [&](size_t bytes) {
    char* p = ws + off;
    off += (bytes + 255) & ~(size_t)255;
    return p;
  };
  const size_t nX = (size_t)B_ * L_ * D_;
  const size_t nW = (size_t)D_ * D_;
  const size_t nE = (size_t)(2 * MAXLEN_ - 1) * HD_;
  const size_t nQ = (size_t)B_ * H_ * L_ * HD_;

  _Float16* Xh   = (_Float16*)alloc(nX * 2);
  _Float16* Wqt  = (_Float16*)alloc(nW * 2);
  _Float16* Wkt  = (_Float16*)alloc(nW * 2);
  _Float16* Wvt  = (_Float16*)alloc(nW * 2);
  _Float16* Wdt  = (_Float16*)alloc(nW * 2);
  _Float16* Eh   = (_Float16*)alloc(nE * 2);
  _Float16* qhp  = (_Float16*)alloc(nQ * 2);
  _Float16* khp  = (_Float16*)alloc(nQ * 2);
  _Float16* vTp  = (_Float16*)alloc(nQ * 2);
  _Float16* ctxh = (_Float16*)alloc(nX * 2);

  k_f32_to_f16<<<(int)((nX + 255) / 256), 256, 0, stream>>>(hidden, Xh, (int)nX);
  k_f32_to_f16<<<(int)((nE + 255) / 256), 256, 0, stream>>>(emb, Eh, (int)nE);

  dim3 tg(D_ / 16, D_ / 16), tb(16, 16);
  k_transpose_f16<<<tg, tb, 0, stream>>>(Wq, Wqt);
  k_transpose_f16<<<tg, tb, 0, stream>>>(Wk, Wkt);
  k_transpose_f16<<<tg, tb, 0, stream>>>(Wv, Wvt);
  k_transpose_f16<<<tg, tb, 0, stream>>>(Wd, Wdt);

  dim3 pg((B_ * L_) / 32, D_ / 64);
  k_proj_qkv<<<pg, 32, 0, stream>>>(Xh, Wqt, bq, qhp, 0);
  k_proj_qkv<<<pg, 32, 0, stream>>>(Xh, Wkt, bk, khp, 0);
  k_proj_qkv<<<pg, 32, 0, stream>>>(Xh, Wvt, bv, vTp, 1);

  // 4 * 12 * 64 = 3072 waves, 4 waves per block
  k_attention<<<768, 128, 0, stream>>>(qhp, khp, vTp, Eh, ctxh);

  k_out_proj<<<pg, 32, 0, stream>>>(ctxh, Wdt, bd, out);
}
